// WindowAttention_20547123544383
// MI455X (gfx1250) — matrix operand order repaired and checked by reading, not compiled, not run on hardware
//
#include <hip/hip_runtime.h>
#include <hip/hip_bf16.h>

typedef __attribute__((ext_vector_type(16))) _Float16 v16h;
typedef __attribute__((ext_vector_type(8)))  float    v8f;
typedef __attribute__((ext_vector_type(4)))  int      v4i;
typedef _Float16 half_t;

#define WMMA_F16(a, b, c) \
  __builtin_amdgcn_wmma_f32_16x16x32_f16(false, (a), false, (b), (short)0, (c), false, false)

#if defined(__gfx1250__) && __has_builtin(__builtin_amdgcn_global_load_async_to_lds_b128) && \
    __has_builtin(__builtin_amdgcn_s_wait_asynccnt)
#define ASYNC_LDS 1
#endif

#ifdef ASYNC_LDS
typedef __attribute__((address_space(1))) v4i gv4i_t;   // global int4
typedef __attribute__((address_space(3))) v4i lv4i_t;   // LDS int4
#endif

// 16-byte global->LDS copy: async (GLOBAL_LOAD_ASYNC_TO_LDS_B128, tracked by
// ASYNCcnt, no VGPR staging) when the builtin exists, else sync copy.
__device__ __forceinline__ void cp16(const half_t* g, half_t* l) {
#ifdef ASYNC_LDS
  __builtin_amdgcn_global_load_async_to_lds_b128((gv4i_t*)g, (lv4i_t*)l, 0, 0);
#else
  *(uint4*)l = *(const uint4*)g;
#endif
}
__device__ __forceinline__ void async_wait_all() {
#ifdef ASYNC_LDS
  __builtin_amdgcn_s_wait_asynccnt(0);
#endif
}

// ---------------- WMMA fragment loaders (wave32, 16x16x32 f16) ----------------
// A 16x32: lane&15 = row M; lane>>4 -> kb in {0,16}; elem e -> K = kb + e
__device__ __forceinline__ v16h frag_a(const half_t* lds, int m0, int k0, int ld) {
  const int lane = threadIdx.x & 31;
  const half_t* p = lds + (m0 + (lane & 15)) * ld + k0 + ((lane >> 4) << 4);
  v16h f;
#pragma unroll
  for (int e = 0; e < 16; ++e) f[e] = p[e];
  return f;
}
// B 32x16 from LDS holding [N][K] rows: lane&15 = col N; lane>>4 -> K half; contiguous K
__device__ __forceinline__ v16h frag_b_nk(const half_t* lds, int n0, int k0, int ld) {
  const int lane = threadIdx.x & 31;
  const half_t* p = lds + (n0 + (lane & 15)) * ld + k0 + ((lane >> 4) << 4);
  v16h f;
#pragma unroll
  for (int e = 0; e < 16; ++e) f[e] = p[e];
  return f;
}
// B 32x16 from LDS holding [K][N]
__device__ __forceinline__ v16h frag_b_kn(const half_t* lds, int n0, int k0, int ld) {
  const int lane = threadIdx.x & 31;
  const int n = n0 + (lane & 15);
  const int kb = k0 + ((lane >> 4) << 4);
  v16h f;
#pragma unroll
  for (int e = 0; e < 16; ++e) f[e] = lds[(kb + e) * ld + n];
  return f;
}

// ---------------- prepass: fused qkv bias (q_bias | 0 | v_bias) ----------------
__global__ void bias_fuse_kernel(const float* __restrict__ qb, const float* __restrict__ vb,
                                 float* __restrict__ bias1536) {
  int i = blockIdx.x * blockDim.x + threadIdx.x;
  if (i >= 1536) return;
  bias1536[i] = (i < 512) ? qb[i] : ((i < 1024) ? 0.f : vb[i - 1024]);
}

// ---------------- prepass: CPB MLP -> tbl(225,16) ----------------
__global__ void rpb_tbl_kernel(const float* __restrict__ table, const float* __restrict__ w1,
                               const float* __restrict__ b1, const float* __restrict__ w2,
                               float* __restrict__ tbl) {
  int gid = blockIdx.x * blockDim.x + threadIdx.x;
  if (gid >= 225 * 16) return;
  int i = gid >> 4, h = gid & 15;
  float c0 = table[i * 2 + 0], c1 = table[i * 2 + 1];
  float acc = 0.f;
  for (int j = 0; j < 512; ++j) {
    float hd = fmaxf(0.f, w1[j * 2 + 0] * c0 + w1[j * 2 + 1] * c1 + b1[j]);
    acc += w2[h * 512 + j] * hd;
  }
  tbl[i * 16 + h] = acc;
}

// ---------------- prepass: expand -> rpb(16,64,64) with 16*sigmoid ----------------
__global__ void rpb_expand_kernel(const float* __restrict__ tbl, const int* __restrict__ idx,
                                  float* __restrict__ rpb) {
  int gid = blockIdx.x * blockDim.x + threadIdx.x;  // 65536
  int h = gid >> 12;
  int r = gid & 4095;
  float x = tbl[idx[r] * 16 + h];
  rpb[gid] = 16.f / (1.f + __expf(-x));
}

// ---------- Kernel: QKV GEMM (131072x512 @ 512x1536) fp32->fp16, double-buffered ----------
__global__ __launch_bounds__(256) void qkv_gemm_kernel(
    const float* __restrict__ x, const float* __restrict__ w,
    const float* __restrict__ bias1536, half_t* __restrict__ qkv) {
  __shared__ alignas(16) half_t As[2][128 * 32];
  __shared__ alignas(16) half_t Bs[2][128 * 32];
  const int tid = threadIdx.x;
  const int wave = tid >> 5;
  const int wm = wave >> 1, wn = wave & 1;
  const int rowBase = blockIdx.y * 128;
  const int colBase = blockIdx.x * 128;

  v8f acc[2][4];
#pragma unroll
  for (int i = 0; i < 2; ++i)
#pragma unroll
    for (int j = 0; j < 4; ++j)
#pragma unroll
      for (int e = 0; e < 8; ++e) acc[i][j][e] = 0.f;

  // prologue: stage K-tile 0
#pragma unroll
  for (int i = 0; i < 4; ++i) {
    int idx4 = tid + i * 256;
    int rr = idx4 >> 3, cc = (idx4 & 7) << 2;
    float4 fa = *(const float4*)(x + (size_t)(rowBase + rr) * 512 + cc);
    float4 fb = *(const float4*)(w + (size_t)(colBase + rr) * 512 + cc);
    half_t* da = As[0] + rr * 32 + cc;
    half_t* db = Bs[0] + rr * 32 + cc;
    da[0] = (half_t)fa.x; da[1] = (half_t)fa.y; da[2] = (half_t)fa.z; da[3] = (half_t)fa.w;
    db[0] = (half_t)fb.x; db[1] = (half_t)fb.y; db[2] = (half_t)fb.z; db[3] = (half_t)fb.w;
  }
  __syncthreads();

  for (int kt = 0; kt < 16; ++kt) {
    const int cur = kt & 1;
    // register-stage next K-tile while computing on current one
    float4 ra[4], rb[4];
    if (kt < 15) {
      const int kofs = (kt + 1) << 5;
#pragma unroll
      for (int i = 0; i < 4; ++i) {
        int idx4 = tid + i * 256;
        int rr = idx4 >> 3, cc = (idx4 & 7) << 2;
        ra[i] = *(const float4*)(x + (size_t)(rowBase + rr) * 512 + kofs + cc);
        rb[i] = *(const float4*)(w + (size_t)(colBase + rr) * 512 + kofs + cc);
        if (kt < 14)
          __builtin_prefetch(x + (size_t)(rowBase + rr) * 512 + kofs + 32 + cc, 0, 1);
      }
    }
    v16h af[2], bf[4];
#pragma unroll
    for (int tm = 0; tm < 2; ++tm) af[tm] = frag_a(As[cur], wm * 32 + tm * 16, 0, 32);
#pragma unroll
    for (int tn = 0; tn < 4; ++tn) bf[tn] = frag_b_nk(Bs[cur], wn * 64 + tn * 16, 0, 32);
#pragma unroll
    for (int tm = 0; tm < 2; ++tm)
#pragma unroll
      for (int tn = 0; tn < 4; ++tn)
        acc[tm][tn] = WMMA_F16(af[tm], bf[tn], acc[tm][tn]);
    if (kt < 15) {
#pragma unroll
      for (int i = 0; i < 4; ++i) {
        int idx4 = tid + i * 256;
        int rr = idx4 >> 3, cc = (idx4 & 7) << 2;
        half_t* da = As[cur ^ 1] + rr * 32 + cc;
        half_t* db = Bs[cur ^ 1] + rr * 32 + cc;
        da[0] = (half_t)ra[i].x; da[1] = (half_t)ra[i].y; da[2] = (half_t)ra[i].z; da[3] = (half_t)ra[i].w;
        db[0] = (half_t)rb[i].x; db[1] = (half_t)rb[i].y; db[2] = (half_t)rb[i].z; db[3] = (half_t)rb[i].w;
      }
    }
    __syncthreads();
  }

  const int lane = tid & 31;
  const int mo = (lane >> 4) << 3, no = lane & 15;
#pragma unroll
  for (int tm = 0; tm < 2; ++tm)
#pragma unroll
    for (int tn = 0; tn < 4; ++tn) {
      int col = colBase + wn * 64 + tn * 16 + no;
      float bv = bias1536[col];
#pragma unroll
      for (int r = 0; r < 8; ++r) {
        int row = rowBase + wm * 32 + tm * 16 + r + mo;
        qkv[(size_t)row * 1536 + col] = (half_t)(acc[tm][tn][r] + bv);
      }
    }
}

// ---------------- Kernel: attention, one wave per (window, head) ----------------
__global__ __launch_bounds__(32) void attn_kernel(
    const half_t* __restrict__ qkv, const float* __restrict__ mask,
    const float* __restrict__ logit_scale, const float* __restrict__ rpb,
    half_t* __restrict__ aout) {
  __shared__ alignas(16) half_t qkbuf[4096];   // qn | kn, later reused as P (64x64 fp16)
  __shared__ alignas(16) half_t vm[64 * 32];
  __shared__ alignas(16) float  Sf[64 * 64];
  half_t* qn = qkbuf;
  half_t* knm = qkbuf + 2048;
  half_t* Pm = qkbuf;

  const int b = blockIdx.x >> 4;
  const int h = blockIdx.x & 15;
  const int lane = threadIdx.x;

  const float scale = __expf(fminf(logit_scale[h], 4.6051702f));   // log(100)
  const float* mk = mask + (size_t)(b & 63) * 4096;
  const float* rp = rpb + (size_t)h * 4096;
  const size_t base = (size_t)b * 64 * 1536 + (size_t)h * 32;

  // kick off V copy (async -> LDS; overlapped with everything until P@V)
#pragma unroll
  for (int i = 0; i < 8; ++i) {
    int ch = lane + i * 32;                 // 256 x 16B chunks
    int t = ch >> 2, cc = (ch & 3) << 3;
    cp16(qkv + base + (size_t)t * 1536 + 1024 + cc, vm + t * 32 + cc);
  }

  // load + l2-normalize q,k (2 token rows per lane)
#pragma unroll
  for (int rr = 0; rr < 2; ++rr) {
    int t = lane + rr * 32;
    const half_t* qr = qkv + base + (size_t)t * 1536;
    const half_t* kr = qr + 512;
    float qs = 0.f, ks = 0.f;
    float qv[32], kv[32];
#pragma unroll
    for (int d = 0; d < 32; ++d) {
      qv[d] = (float)qr[d]; qs += qv[d] * qv[d];
      kv[d] = (float)kr[d]; ks += kv[d] * kv[d];
    }
    float qi = 1.f / fmaxf(sqrtf(qs), 1e-12f);
    float ki = 1.f / fmaxf(sqrtf(ks), 1e-12f);
#pragma unroll
    for (int d = 0; d < 32; ++d) {
      qn[t * 32 + d] = (half_t)(qv[d] * qi);
      knm[t * 32 + d] = (half_t)(kv[d] * ki);
    }
  }
  __syncthreads();

  // S = Qn @ Kn^T  (4x4 tiles, one WMMA each)
  v8f S[4][4];
#pragma unroll
  for (int i = 0; i < 4; ++i)
#pragma unroll
    for (int j = 0; j < 4; ++j)
#pragma unroll
      for (int e = 0; e < 8; ++e) S[i][j][e] = 0.f;
  v16h qf[4], kf[4];
#pragma unroll
  for (int tm = 0; tm < 4; ++tm) qf[tm] = frag_a(qn, tm * 16, 0, 32);
#pragma unroll
  for (int tn = 0; tn < 4; ++tn) kf[tn] = frag_b_nk(knm, tn * 16, 0, 32);
#pragma unroll
  for (int tm = 0; tm < 4; ++tm)
#pragma unroll
    for (int tn = 0; tn < 4; ++tn)
      S[tm][tn] = WMMA_F16(qf[tm], kf[tn], S[tm][tn]);

  // epilogue: scale + rel-pos bias + mask -> Sf
  const int mo = (lane >> 4) << 3, no = lane & 15;
#pragma unroll
  for (int tm = 0; tm < 4; ++tm)
#pragma unroll
    for (int tn = 0; tn < 4; ++tn)
#pragma unroll
      for (int r = 0; r < 8; ++r) {
        int m = tm * 16 + r + mo, n = tn * 16 + no;
        Sf[m * 64 + n] = S[tm][tn][r] * scale + rp[m * 64 + n] + mk[m * 64 + n];
      }
  __syncthreads();

  // softmax (2 rows per lane) -> Pm fp16 (reuses q/k LDS)
#pragma unroll
  for (int rr = 0; rr < 2; ++rr) {
    int m = lane + rr * 32;
    float mx = -1e30f;
    for (int n = 0; n < 64; ++n) mx = fmaxf(mx, Sf[m * 64 + n]);
    float s = 0.f;
    for (int n = 0; n < 64; ++n) {
      float e = __expf(Sf[m * 64 + n] - mx);
      Sf[m * 64 + n] = e; s += e;
    }
    float inv = 1.f / (s + 1e-12f);
    for (int n = 0; n < 64; ++n) Pm[m * 64 + n] = (half_t)(Sf[m * 64 + n] * inv);
  }
  async_wait_all();    // V must be resident before P@V
  __syncthreads();

  // O = P @ V  (4x2 tiles, K=64 -> 2 WMMA chunks)
  v8f O[4][2];
#pragma unroll
  for (int i = 0; i < 4; ++i)
#pragma unroll
    for (int j = 0; j < 2; ++j)
#pragma unroll
      for (int e = 0; e < 8; ++e) O[i][j][e] = 0.f;
#pragma unroll
  for (int kc = 0; kc < 2; ++kc) {
    v16h pf[4], vf[2];
#pragma unroll
    for (int tm = 0; tm < 4; ++tm) pf[tm] = frag_a(Pm, tm * 16, kc * 32, 64);
#pragma unroll
    for (int tn = 0; tn < 2; ++tn) vf[tn] = frag_b_kn(vm, tn * 16, kc * 32, 32);
#pragma unroll
    for (int tm = 0; tm < 4; ++tm)
#pragma unroll
      for (int tn = 0; tn < 2; ++tn)
        O[tm][tn] = WMMA_F16(pf[tm], vf[tn], O[tm][tn]);
  }
#pragma unroll
  for (int tm = 0; tm < 4; ++tm)
#pragma unroll
    for (int tn = 0; tn < 2; ++tn)
#pragma unroll
      for (int r = 0; r < 8; ++r) {
        int m = tm * 16 + r + mo, d = tn * 16 + no;
        aout[((size_t)b * 64 + m) * 512 + h * 32 + d] = (half_t)O[tm][tn][r];
      }
}

// ------ Kernel: proj GEMM (131072x512 @ 512x512) + bias -> fp32, async A tiles ------
__global__ __launch_bounds__(256) void proj_gemm_kernel(
    const half_t* __restrict__ a, const float* __restrict__ w,
    const float* __restrict__ bias, float* __restrict__ out) {
  __shared__ alignas(16) half_t As[2][128 * 32];
  __shared__ alignas(16) half_t Bs[2][128 * 32];
  const int tid = threadIdx.x;
  const int wave = tid >> 5;
  const int wm = wave >> 1, wn = wave & 1;
  const int rowBase = blockIdx.y * 128;
  const int colBase = blockIdx.x * 128;

  v8f acc[2][4];
#pragma unroll
  for (int i = 0; i < 2; ++i)
#pragma unroll
    for (int j = 0; j < 4; ++j)
#pragma unroll
      for (int e = 0; e < 8; ++e) acc[i][j][e] = 0.f;

  // prologue: A tile 0 async, B tile 0 through registers (needs fp32->fp16)
#pragma unroll
  for (int i = 0; i < 2; ++i) {
    int ch = tid + i * 256;
    int rr = ch >> 2, cc = (ch & 3) << 3;
    cp16(a + (size_t)(rowBase + rr) * 512 + cc, As[0] + rr * 32 + cc);
  }
#pragma unroll
  for (int i = 0; i < 4; ++i) {
    int idx4 = tid + i * 256;
    int rr = idx4 >> 3, cc = (idx4 & 7) << 2;
    float4 fb = *(const float4*)(w + (size_t)(colBase + rr) * 512 + cc);
    half_t* db = Bs[0] + rr * 32 + cc;
    db[0] = (half_t)fb.x; db[1] = (half_t)fb.y; db[2] = (half_t)fb.z; db[3] = (half_t)fb.w;
  }
  async_wait_all();
  __syncthreads();

  for (int kt = 0; kt < 16; ++kt) {
    const int cur = kt & 1;
    float4 rb[4];
    if (kt < 15) {
      const int kofs = (kt + 1) << 5;
#pragma unroll
      for (int i = 0; i < 2; ++i) {       // next A tile: async straight to LDS
        int ch = tid + i * 256;
        int rr = ch >> 2, cc = (ch & 3) << 3;
        cp16(a + (size_t)(rowBase + rr) * 512 + kofs + cc, As[cur ^ 1] + rr * 32 + cc);
      }
#pragma unroll
      for (int i = 0; i < 4; ++i) {       // next B tile: registers (convert)
        int idx4 = tid + i * 256;
        int rr = idx4 >> 3, cc = (idx4 & 7) << 2;
        rb[i] = *(const float4*)(w + (size_t)(colBase + rr) * 512 + kofs + cc);
        if (kt < 14)
          __builtin_prefetch(a + (size_t)(rowBase + rr) * 512 + kofs + 32 + cc, 0, 1);
      }
    }
    v16h af[2], bf[4];
#pragma unroll
    for (int tm = 0; tm < 2; ++tm) af[tm] = frag_a(As[cur], wm * 32 + tm * 16, 0, 32);
#pragma unroll
    for (int tn = 0; tn < 4; ++tn) bf[tn] = frag_b_nk(Bs[cur], wn * 64 + tn * 16, 0, 32);
#pragma unroll
    for (int tm = 0; tm < 2; ++tm)
#pragma unroll
      for (int tn = 0; tn < 4; ++tn)
        acc[tm][tn] = WMMA_F16(af[tm], bf[tn], acc[tm][tn]);
    if (kt < 15) {
#pragma unroll
      for (int i = 0; i < 4; ++i) {
        int idx4 = tid + i * 256;
        int rr = idx4 >> 3, cc = (idx4 & 7) << 2;
        half_t* db = Bs[cur ^ 1] + rr * 32 + cc;
        db[0] = (half_t)rb[i].x; db[1] = (half_t)rb[i].y; db[2] = (half_t)rb[i].z; db[3] = (half_t)rb[i].w;
      }
    }
    async_wait_all();
    __syncthreads();
  }

  const int lane = tid & 31;
  const int mo = (lane >> 4) << 3, no = lane & 15;
#pragma unroll
  for (int tm = 0; tm < 2; ++tm)
#pragma unroll
    for (int tn = 0; tn < 4; ++tn) {
      int col = colBase + wn * 64 + tn * 16 + no;
      float bv = bias[col];
#pragma unroll
      for (int r = 0; r < 8; ++r) {
        int row = rowBase + wm * 32 + tm * 16 + r + mo;
        out[(size_t)row * 512 + col] = acc[tm][tn][r] + bv;
      }
    }
}

extern "C" void kernel_launch(void* const* d_in, const int* in_sizes, int n_in,
                              void* d_out, int out_size, void* d_ws, size_t ws_size,
                              hipStream_t stream) {
  const float* x           = (const float*)d_in[0];
  const float* mask        = (const float*)d_in[1];
  const float* qkv_w       = (const float*)d_in[2];
  const float* q_bias      = (const float*)d_in[3];
  const float* v_bias      = (const float*)d_in[4];
  const float* logit_scale = (const float*)d_in[5];
  const float* cpb_w1      = (const float*)d_in[6];
  const float* cpb_b1      = (const float*)d_in[7];
  const float* cpb_w2      = (const float*)d_in[8];
  const float* proj_w      = (const float*)d_in[9];
  const float* proj_b      = (const float*)d_in[10];
  const float* table       = (const float*)d_in[11];
  const int*   idx         = (const int*)d_in[12];
  float* out = (float*)d_out;

  char* ws = (char*)d_ws;
  const size_t ROWS = (size_t)2048 * 64;               // 131072 tokens
  const size_t QKV_BYTES  = ROWS * 1536 * sizeof(half_t);
  const size_t AOUT_BYTES = ROWS * 512 * sizeof(half_t);
  half_t* qkv  = (half_t*)ws;
  half_t* aout = (half_t*)(ws + QKV_BYTES);
  float*  tbl  = (float*)(ws + QKV_BYTES + AOUT_BYTES);
  float*  rpb  = (float*)(ws + QKV_BYTES + AOUT_BYTES + 16384);
  float*  bias1536 = (float*)(ws + QKV_BYTES + AOUT_BYTES + 16384 + 262144);

  bias_fuse_kernel<<<6, 256, 0, stream>>>(q_bias, v_bias, bias1536);
  rpb_tbl_kernel<<<15, 256, 0, stream>>>(table, cpb_w1, cpb_b1, cpb_w2, tbl);
  rpb_expand_kernel<<<256, 256, 0, stream>>>(tbl, idx, rpb);
  qkv_gemm_kernel<<<dim3(12, 1024), 256, 0, stream>>>(x, qkv_w, bias1536, qkv);
  attn_kernel<<<2048 * 16, 32, 0, stream>>>(qkv, mask, logit_scale, rpb, aout);
  proj_gemm_kernel<<<dim3(4, 1024), 256, 0, stream>>>(aout, proj_w, proj_b, out);
}
